// VectorQuantizer_38843684225126
// MI455X (gfx1250) — matrix operand order repaired and checked by reading, not compiled, not run on hardware
//
#include <hip/hip_runtime.h>

// ---- CDNA5 WMMA types ----
typedef __attribute__((ext_vector_type(16))) __bf16       v16bf;
typedef __attribute__((ext_vector_type(8)))  float        v8f;
typedef __attribute__((ext_vector_type(8)))  unsigned int v8u;

#define NUM_K  512      // codebook entries
#define CDIM   64       // embedding dim
#define HW     4096     // 64*64 spatial
#define NBATCH 32
#define TILE_N 128      // z rows per workgroup
#define KTILE  256      // codebook rows staged in LDS at a time
#define CBS    70       // bf16 stride for codebook LDS rows (35 dwords, odd -> no bank conflicts)
#define ZSTR   70       // bf16 stride for z LDS rows

static __device__ __forceinline__ unsigned short f2bf(float f) {
    unsigned u = __builtin_bit_cast(unsigned, f);
    u += 0x7FFFu + ((u >> 16) & 1u);        // round-to-nearest-even
    return (unsigned short)(u >> 16);
}

__global__ __launch_bounds__(256) void vq_wmma_kernel(
    const float* __restrict__ z,        // [32][64][64][64]  (B,C,H,W)
    const float* __restrict__ cb,       // [512][64]
    float* __restrict__ zq,             // [32][64][64][64]
    float* __restrict__ idx_out)        // [131072] (indices as float)
{
    __shared__ __align__(16) unsigned short s_cb[KTILE * CBS];   // 35840 B
    __shared__ __align__(16) unsigned short s_z [TILE_N * ZSTR]; // 17920 B
    __shared__ float s_cn[NUM_K];                                //  2048 B
    __shared__ int   s_idx[TILE_N];                              //   512 B

    const int t    = threadIdx.x;
    const int wave = t >> 5;
    const int lane = t & 31;
    const int half = lane >> 4;     // lane group 0-15 / 16-31
    const int lm   = lane & 15;

    const int n0  = blockIdx.x * TILE_N;     // first flat n of this tile
    const int b   = n0 >> 12;                // n0 / HW
    const int hw0 = n0 & (HW - 1);

    const float* zg = z + (size_t)b * CDIM * HW + hw0;

    // ---- stage z tile: [128 n][64 c] bf16, transposed from C-major global ----
    #pragma unroll
    for (int it = 0; it < 32; ++it) {
        int lin = it * 256 + t;
        int c = lin >> 7;              // 0..63
        int j = lin & 127;             // 0..127
        s_z[j * ZSTR + c] = f2bf(zg[c * HW + j]);     // coalesced global read
    }
    // ---- stage codebook tile 0 as bf16 ----
    #pragma unroll 4
    for (int it = 0; it < 64; ++it) {
        int lin = it * 256 + t;
        int kl = lin >> 6;
        int c  = lin & 63;
        s_cb[kl * CBS + c] = f2bf(cb[kl * CDIM + c]);
    }
    // ---- exact fp32 codebook norms ||c_k||^2 ----
    for (int k = t; k < NUM_K; k += 256) {
        const float* row = cb + k * CDIM;
        float s = 0.f;
        #pragma unroll
        for (int c = 0; c < CDIM; ++c) { float x = row[c]; s = fmaf(x, x, s); }
        s_cn[k] = s;
    }
    __syncthreads();

    // ---- load A fragments (this wave's 16 z rows), reused for all 512 k ----
    const unsigned* z32 = (const unsigned*)s_z;
    const int row = wave * 16 + lm;                 // A row m = lane&15
    v8u a0u, a1u;
    #pragma unroll
    for (int v = 0; v < 8; ++v) {
        // 16-bit A 16x32 layout: Kpair base = (v>>2)*16 + half*8 + (v&3)*2
        int w = row * 35 + ((v >> 2) * 8 + half * 4 + (v & 3));
        a0u[v] = z32[w];        // c chunk 0..31
        a1u[v] = z32[w + 16];   // c chunk 32..63
    }
    const v16bf a0 = __builtin_bit_cast(v16bf, a0u);
    const v16bf a1 = __builtin_bit_cast(v16bf, a1u);

    float bd[8];
    int   bi[8];
    #pragma unroll
    for (int v = 0; v < 8; ++v) { bd[v] = 3.4e38f; bi[v] = 0; }

    const unsigned* c32 = (const unsigned*)s_cb;

    for (int kt = 0; kt < NUM_K / KTILE; ++kt) {
        if (kt > 0) {
            __syncthreads();
            #pragma unroll 4
            for (int it = 0; it < 64; ++it) {
                int lin = it * 256 + t;
                int kl = lin >> 6;
                int c  = lin & 63;
                s_cb[kl * CBS + c] = f2bf(cb[(kt * KTILE + kl) * CDIM + c]);
            }
            __syncthreads();
        }
        // column this lane owns within each 16-wide k-tile
        #pragma unroll 4
        for (int kk0 = 0; kk0 < KTILE; kk0 += 16) {
            const int kk = kk0 + lm;                     // local codebook row
            // B fragment: lane holds column n=lm; VGPR v = K pair (v>>2)*16 + half*8 + (v&3)*2
            // => LDS dwords at kk*35 + chunk*16 + half*4 + (v>>2)*8 + (v&3)
            const int baseB = kk * 35 + half * 4;
            v8u b0u, b1u;
            #pragma unroll
            for (int v = 0; v < 8; ++v) {
                b0u[v] = c32[baseB + v + (v >> 2) * 4];          // c chunk 0..31
                b1u[v] = c32[baseB + 16 + v + (v >> 2) * 4];     // c chunk 32..63
            }
            v8f acc;
            #pragma unroll
            for (int v = 0; v < 8; ++v) acc[v] = 0.f;
            acc = __builtin_amdgcn_wmma_f32_16x16x32_bf16(
                    false, a0, false, __builtin_bit_cast(v16bf, b0u),
                    (short)0, acc, false, false);
            acc = __builtin_amdgcn_wmma_f32_16x16x32_bf16(
                    false, a1, false, __builtin_bit_cast(v16bf, b1u),
                    (short)0, acc, false, false);

            const int kglob = kt * KTILE + kk;           // lane's column index
            const float cn = s_cn[kglob];
            #pragma unroll
            for (int v = 0; v < 8; ++v) {
                float d = fmaf(-2.f, acc[v], cn);        // ||c||^2 - 2 z.c
                if (d < bd[v]) { bd[v] = d; bi[v] = kglob; }
            }
        }
    }

    // ---- reduce (min,argmin) across the 16 lanes sharing each row ----
    #pragma unroll
    for (int v = 0; v < 8; ++v) {
        #pragma unroll
        for (int m = 1; m < 16; m <<= 1) {
            float od = __shfl_xor(bd[v], m, 32);
            int   oi = __shfl_xor(bi[v], m, 32);
            if (od < bd[v] || (od == bd[v] && oi < bi[v])) { bd[v] = od; bi[v] = oi; }
        }
        if (lm == 0) s_idx[wave * 16 + half * 8 + v] = bi[v];  // row m = v + 8*half
    }
    __syncthreads();

    // ---- outputs ----
    if (t < TILE_N) idx_out[n0 + t] = (float)s_idx[t];

    float* zqg = zq + (size_t)b * CDIM * HW + hw0;
    #pragma unroll
    for (int it = 0; it < 32; ++it) {
        int lin = it * 256 + t;
        int c = lin >> 7;
        int j = lin & 127;
        zqg[c * HW + j] = cb[s_idx[j] * CDIM + c];   // exact fp32 codebook gather
    }
}

extern "C" void kernel_launch(void* const* d_in, const int* in_sizes, int n_in,
                              void* d_out, int out_size, void* d_ws, size_t ws_size,
                              hipStream_t stream) {
    const float* z_e      = (const float*)d_in[0];   // [32,64,64,64] fp32
    const float* codebook = (const float*)d_in[1];   // [512,64] fp32
    float* zq      = (float*)d_out;                  // first output, 8388608 floats
    float* idx_out = (float*)d_out + (size_t)NBATCH * CDIM * HW;  // 131072 indices

    const int num_blocks = (NBATCH * HW) / TILE_N;   // 1024
    vq_wmma_kernel<<<num_blocks, 256, 0, stream>>>(z_e, codebook, zq, idx_out);
}
